// LastQueryTransformerEncoderLayer_54941221650534
// MI455X (gfx1250) — matrix operand order repaired and checked by reading, not compiled, not run on hardware
//
#include <hip/hip_runtime.h>
#include <hip/hip_bf16.h>

// ---------------------------------------------------------------------------
// LastQueryTransformerEncoderLayer for MI455X (gfx1250, wave32, WMMA + TDM).
//
// Algebraic restructuring (single query per (b,h)):
//   scores[b,h,s] = src[s,b,:] . qW[b,h,:],  qW[b,h,:] = Wk_h^T q[b,h,:]
//   ctx[b,h,:]    = Wv_h @ y[b,h,:],         y[b,h,:]  = sum_s attn[b,h,s]*src[s,b,:]
// -> K/V (each 256 MB) never materialized; total math ~2.2 GFLOP;
//    traffic ~1 GB -> bandwidth bound (~44us @ 23.3 TB/s).
// GEMMs: v_wmma_f32_16x16x32_bf16 (fp32 accumulate).
// y-pass strided gather staged through LDS via TENSOR_LOAD_TO_LDS (TDM),
// double buffered, s_wait_tensorcnt synchronized, K split 8x for parallelism.
// ---------------------------------------------------------------------------

typedef __attribute__((ext_vector_type(16))) __bf16 v16bf;
typedef __attribute__((ext_vector_type(8)))  float  v8f;
typedef __attribute__((ext_vector_type(4)))  unsigned int u32x4;
typedef __attribute__((ext_vector_type(8)))  int i32x8;
typedef __attribute__((ext_vector_type(4)))  int i32x4;

#define DEV static __device__ __forceinline__

constexpr int S_ = 2048, B_ = 64, D_ = 512, H_ = 8, HD_ = 64;
constexpr int KSPLIT_ = 8;                 // y-pass K split factor
constexpr float SCALE_ = 0.125f;           // 1/sqrt(64)
constexpr float LNEPS_ = 1e-5f;

// workspace layout (float offsets)
constexpr size_t WS_Q   = 0;                                          // B*D
constexpr size_t WS_QW  = WS_Q   + (size_t)B_ * D_;                   // B*H*D
constexpr size_t WS_SC  = WS_QW  + (size_t)B_ * H_ * D_;              // B*H*S
constexpr size_t WS_Y   = WS_SC  + (size_t)B_ * H_ * S_;              // B*H*D
constexpr size_t WS_YP  = WS_Y   + (size_t)B_ * H_ * D_;              // KSPLIT*B*H*D
constexpr size_t WS_CTX = WS_YP  + (size_t)KSPLIT_ * B_ * H_ * D_;    // B*D
constexpr size_t WS_AO  = WS_CTX + (size_t)B_ * D_;                   // B*D

// --- WMMA operand loaders (fp32 memory -> bf16 lane registers) --------------
// A 16x32 bf16: lane<16 holds row m=lane, K=k0..k0+15;
// lane>=16 same row, K=k0+16..k0+31.
DEV v16bf loadA_row(const float* __restrict__ row, int k0, int lane) {
  const float* p = row + k0 + ((lane & 16) ? 16 : 0);
  v16bf r;
#pragma unroll
  for (int i = 0; i < 16; ++i) {
    r[i]     = (__bf16)p[i];
  }
  return r;
}
// B 32x16 bf16: lane = n; lane<16 holds K=k0..k0+15, lane>=16 K=k0+16..k0+31.
// Source row-major W with B[k][n] = W[n][k] -> contiguous along k in row n.
DEV v16bf loadB_row(const float* __restrict__ rowN, int k0, int lane) {
  const float* p = rowN + k0 + ((lane & 16) ? 16 : 0);
  v16bf r;
#pragma unroll
  for (int i = 0; i < 16; ++i) r[i] = (__bf16)p[i];
  return r;
}
// B[k][n] stored row-major along n (stride ld between k rows).
DEV v16bf loadB_strided(const float* __restrict__ base, long ld, int col, int k0, int lane) {
  const float* p = base + (long)(k0 + ((lane & 16) ? 16 : 0)) * ld + col;
  v16bf r;
#pragma unroll
  for (int i = 0; i < 16; ++i) r[i] = (__bf16)p[(long)i * ld];
  return r;
}
DEV v8f wmma_bf16(v16bf a, v16bf b, v8f c) {
  return __builtin_amdgcn_wmma_f32_16x16x32_bf16(false, a, false, b, (short)0, c, false, false);
}

// --- TDM: async DMA of a 2D fp32 tile (tile_d contiguous x tile_s strided
//     rows) from global memory into LDS, packed contiguously (row-major,
//     d fastest). D# per CDNA5 ISA.
DEV void tdm_load_2d(unsigned int lds_off, const void* gptr,
                     unsigned int tile_d, unsigned int tile_s,
                     unsigned int row_stride_elems) {
  const unsigned long long ga = (unsigned long long)gptr;
  u32x4 g0;
  g0[0] = 1u;                                          // count=1, user mode
  g0[1] = lds_off;                                     // lds_addr (bytes)
  g0[2] = (unsigned int)(ga & 0xFFFFFFFFull);          // global_addr[31:0]
  g0[3] = (unsigned int)((ga >> 32) & 0x1FFFFFFull)    // global_addr[56:32]
        | (2u << 30);                                  // type=2 ("image")
  i32x8 g1;
  g1[0] = (int)(2u << 16);                             // data_size=4B; mask=0
  g1[1] = (int)(tile_d << 16);                         // tensor_dim0[15:0]
  g1[2] = (int)(tile_s << 16);                         // dim0 hi=0 | tensor_dim1 lo
  g1[3] = (int)(tile_d << 16);                         // dim1 hi=0 | tile_dim0
  g1[4] = (int)tile_s;                                 // tile_dim1 | tile_dim2=0
  g1[5] = (int)row_stride_elems;                       // tensor_dim0_stride[31:0]
  g1[6] = 0;                                           // stride hi | dim1_stride lo
  g1[7] = 0;
  const i32x4 gz4 = {0, 0, 0, 0};                      // 2D: groups 2/3 unused
  const i32x8 gz8 = {0, 0, 0, 0, 0, 0, 0, 0};
  __builtin_amdgcn_tensor_load_to_lds(g0, g1, gz4, gz4, gz8, 0);
}

// --- K1: q = (gather(src) @ Wq^T + bq) * scale  (64x512, K=512) -------------
__global__ void q_proj_kernel(const float* __restrict__ src, const int* __restrict__ seq,
                              const float* __restrict__ ipw, const float* __restrict__ ipb,
                              float* __restrict__ ws) {
  const int lane = threadIdx.x;
  const int nt = blockIdx.x, mt = blockIdx.y;
  const int br = mt * 16 + (lane & 15);
  const int nr = nt * 16 + (lane & 15);
  const float* arow = src + ((size_t)(seq[br] - 1) * B_ + br) * D_;
  const float* brow = ipw + (size_t)nr * D_;              // Wq row nr
  v8f acc = {};
  for (int k0 = 0; k0 < D_; k0 += 32)
    acc = wmma_bf16(loadA_row(arow, k0, lane), loadB_row(brow, k0, lane), acc);
  float* q = ws + WS_Q;
#pragma unroll
  for (int i = 0; i < 8; ++i) {
    const int m = mt * 16 + i + ((lane >> 4) << 3);
    q[(size_t)m * D_ + nr] = (acc[i] + ipb[nr]) * SCALE_;
  }
}

// --- K2: qW[b,h,d] = sum_j q[b,h*64+j] * Wk[h*64+j, d]  (per-h 64x512, K=64)
__global__ void qw_kernel(const float* __restrict__ ipw, float* __restrict__ ws) {
  const int lane = threadIdx.x;
  const int nt = blockIdx.x, mt = blockIdx.y, h = blockIdx.z;
  const float* q = ws + WS_Q;
  const int br = mt * 16 + (lane & 15);
  const float* arow  = q + (size_t)br * D_ + h * HD_;
  const float* bbase = ipw + (size_t)(D_ + h * HD_) * D_;  // Wk_h: 64 x D
  const int n = nt * 16 + (lane & 15);
  v8f acc = {};
  for (int k0 = 0; k0 < HD_; k0 += 32)
    acc = wmma_bf16(loadA_row(arow, k0, lane), loadB_strided(bbase, D_, n, k0, lane), acc);
  float* qw = ws + WS_QW;
#pragma unroll
  for (int i = 0; i < 8; ++i) {
    const int m = mt * 16 + i + ((lane >> 4) << 3);
    qw[((size_t)m * H_ + h) * D_ + n] = acc[i];
  }
}

// --- K3: scores[b,h,s] = src[s,b,:] . qW[b,h,:]  (per-b S x 8pad16, K=512) --
__global__ void scores_kernel(const float* __restrict__ src, float* __restrict__ ws) {
  const int lane = threadIdx.x;
  const int st = blockIdx.x;          // s-tile
  const int b  = blockIdx.y;
  const int srow = st * 16 + (lane & 15);
  const float* arow = src + ((size_t)srow * B_ + b) * D_;
  const float* qw = ws + WS_QW;
  const int hcol = (lane & 15) & 7;   // pad N=8 -> 16, dup rows, discard on store
  const float* brow = qw + ((size_t)b * H_ + hcol) * D_;
  v8f acc = {};
  for (int k0 = 0; k0 < D_; k0 += 32)
    acc = wmma_bf16(loadA_row(arow, k0, lane), loadB_row(brow, k0, lane), acc);
  float* sc = ws + WS_SC;
  const int n = lane & 15;
  if (n < H_) {
#pragma unroll
    for (int i = 0; i < 8; ++i) {
      const int sidx = st * 16 + i + ((lane >> 4) << 3);
      sc[((size_t)b * H_ + n) * S_ + sidx] = acc[i];
    }
  }
}

// --- K4: softmax over S for each of B*H rows --------------------------------
__global__ void softmax_kernel(float* __restrict__ ws) {
  float* row = ws + WS_SC + (size_t)blockIdx.x * S_;
  const int t = threadIdx.x;          // 256 threads, 8 elems each
  const int w = t >> 5, l = t & 31;
  __shared__ float redm[8], reds[8];
  float v[8], m = -3.402823466e38f;
#pragma unroll
  for (int i = 0; i < 8; ++i) { v[i] = row[t + i * 256]; m = fmaxf(m, v[i]); }
#pragma unroll
  for (int off = 16; off > 0; off >>= 1) m = fmaxf(m, __shfl_xor(m, off, 32));
  if (l == 0) redm[w] = m;
  __syncthreads();
  float bm = redm[0];
#pragma unroll
  for (int i = 1; i < 8; ++i) bm = fmaxf(bm, redm[i]);
  float s = 0.f;
#pragma unroll
  for (int i = 0; i < 8; ++i) { v[i] = __expf(v[i] - bm); s += v[i]; }
#pragma unroll
  for (int off = 16; off > 0; off >>= 1) s += __shfl_xor(s, off, 32);
  if (l == 0) reds[w] = s;
  __syncthreads();
  float bs = 0.f;
#pragma unroll
  for (int i = 0; i < 8; ++i) bs += reds[i];
  const float inv = 1.f / bs;
#pragma unroll
  for (int i = 0; i < 8; ++i) row[t + i * 256] = v[i] * inv;
}

// --- K5 helper: one WMMA step consuming a 32x16 LDS tile --------------------
DEV void consume_tile(const float* __restrict__ lb, const float* __restrict__ arow,
                      int j, int lane, v8f& acc) {
  const int col = lane & 15;
  const int ks = (lane & 16) ? 16 : 0;
  v16bf bb;
#pragma unroll
  for (int i = 0; i < 16; ++i) bb[i] = (__bf16)lb[(ks + i) * 16 + col];
  acc = wmma_bf16(loadA_row(arow, j * 32, lane), bb, acc);
}

// --- K5: y partials via TDM-staged tiles ------------------------------------
// y[b,h,d] = sum_s attn[b,h,s]*src[s,b,d]; K=2048 split KSPLIT ways.
// Each wave owns a 16-wide d tile; B operand (16 d x 32 s, stride B*D) is
// DMA'd into LDS by the Tensor Data Mover, double buffered.
__global__ void __launch_bounds__(32) y_kernel(const float* __restrict__ src,
                                               float* __restrict__ ws) {
  const int lane = threadIdx.x;
  const int dt = blockIdx.x;          // d-tile 0..31
  const int b  = blockIdx.y;          // 0..63
  const int kc = blockIdx.z;          // k-chunk 0..KSPLIT-1
  __shared__ float lbuf[2][32 * 16];  // two 2KB tiles
  const int kbase = kc * (S_ / KSPLIT_);
  const int dcol0 = dt * 16;
  const float* sc = ws + WS_SC;
  const int mrow = (lane & 15) & 7;   // pad M=8 -> 16
  const float* arow = sc + ((size_t)b * H_ + mrow) * S_ + kbase;
  const int dcol = dcol0 + (lane & 15);
  const unsigned int lds0 = (unsigned int)(size_t)(void*)&lbuf[0][0];
  const unsigned int lds1 = (unsigned int)(size_t)(void*)&lbuf[1][0];

  // prologue: DMA tile 0
  tdm_load_2d(lds0, src + ((size_t)kbase * B_ + b) * D_ + dcol0, 16, 32, B_ * D_);
  v8f acc = {};
#pragma unroll
  for (int j = 0; j < 7; ++j) {
    // DMA next tile into the other buffer (tile t lives in buffer t&1)
    tdm_load_2d(((j + 1) & 1) ? lds1 : lds0,
                src + ((size_t)(kbase + (j + 1) * 32) * B_ + b) * D_ + dcol0,
                16, 32, B_ * D_);
    __builtin_amdgcn_s_wait_tensorcnt(1);   // tile j complete
    consume_tile((j & 1) ? &lbuf[1][0] : &lbuf[0][0], arow, j, lane, acc);
  }
  __builtin_amdgcn_s_wait_tensorcnt(0);     // last tile complete
  consume_tile(&lbuf[1][0], arow, 7, lane, acc);   // tile 7 -> buffer 1

  float* yp = ws + WS_YP + (size_t)kc * ((size_t)B_ * H_ * D_);
#pragma unroll
  for (int i = 0; i < 8; ++i) {
    const int m = i + ((lane >> 4) << 3);
    if (m < H_) yp[((size_t)b * H_ + m) * D_ + dcol] = acc[i];
  }
}

// --- K5b: reduce y partials --------------------------------------------------
__global__ void y_reduce_kernel(float* __restrict__ ws) {
  const size_t i = (size_t)blockIdx.x * blockDim.x + threadIdx.x;  // < B*H*D
  const float* yp = ws + WS_YP;
  float s = 0.f;
#pragma unroll
  for (int k = 0; k < KSPLIT_; ++k) s += yp[(size_t)k * ((size_t)B_ * H_ * D_) + i];
  ws[WS_Y + i] = s;
}

// --- K6: ctx[b, h*64+j] = sum_d Wv[h*64+j, d] * y[b,h,d]  (per-h 64x64) -----
__global__ void ctx_kernel(const float* __restrict__ ipw, float* __restrict__ ws) {
  const int lane = threadIdx.x;
  const int nt = blockIdx.x, mt = blockIdx.y, h = blockIdx.z;
  const float* y = ws + WS_Y;
  const int br = mt * 16 + (lane & 15);
  const float* arow = y + ((size_t)br * H_ + h) * D_;
  const int n = nt * 16 + (lane & 15);
  const float* brow = ipw + (size_t)(2 * D_ + h * HD_ + n) * D_;   // Wv row
  v8f acc = {};
  for (int k0 = 0; k0 < D_; k0 += 32)
    acc = wmma_bf16(loadA_row(arow, k0, lane), loadB_row(brow, k0, lane), acc);
  float* ctx = ws + WS_CTX;
#pragma unroll
  for (int i = 0; i < 8; ++i) {
    const int m = mt * 16 + i + ((lane >> 4) << 3);
    ctx[(size_t)m * D_ + h * HD_ + n] = acc[i];
  }
}

// --- K7: attn_out = ctx @ Wo^T + bo  (64x512, K=512) ------------------------
__global__ void oproj_kernel(const float* __restrict__ opw, const float* __restrict__ opb,
                             float* __restrict__ ws) {
  const int lane = threadIdx.x;
  const int nt = blockIdx.x, mt = blockIdx.y;
  const float* ctx = ws + WS_CTX;
  const float* arow = ctx + (size_t)(mt * 16 + (lane & 15)) * D_;
  const int n = nt * 16 + (lane & 15);
  const float* brow = opw + (size_t)n * D_;
  v8f acc = {};
  for (int k0 = 0; k0 < D_; k0 += 32)
    acc = wmma_bf16(loadA_row(arow, k0, lane), loadB_row(brow, k0, lane), acc);
  float* ao = ws + WS_AO;
#pragma unroll
  for (int i = 0; i < 8; ++i) {
    const int m = mt * 16 + i + ((lane >> 4) << 3);
    ao[(size_t)m * D_ + n] = acc[i] + opb[n];
  }
}

// --- K8: out = LayerNorm(src + attn_out)  — 1 wave32 per row of 512 ---------
__global__ void ln_kernel(const float* __restrict__ src, const float* __restrict__ lw,
                          const float* __restrict__ lb, const float* __restrict__ ws,
                          float* __restrict__ out) {
  const int w = threadIdx.x >> 5, lane = threadIdx.x & 31;
  const size_t r = (size_t)blockIdx.x * 8 + w;     // row index over S*B
  const int b = (int)(r & (B_ - 1));
  const float* ao = ws + WS_AO + (size_t)b * D_;
  const float* x0 = src + r * D_;
  if (r + 8 < (size_t)S_ * B_)                     // stream hint (speculative)
    __builtin_prefetch(x0 + (size_t)8 * D_ + lane * 16, 0, 0);
  float x[16];
  float s = 0.f;
#pragma unroll
  for (int j = 0; j < 4; ++j) {
    const float4 sv = *(const float4*)(x0 + lane * 16 + j * 4);
    const float4 av = *(const float4*)(ao + lane * 16 + j * 4);
    x[j*4+0] = sv.x + av.x;  x[j*4+1] = sv.y + av.y;
    x[j*4+2] = sv.z + av.z;  x[j*4+3] = sv.w + av.w;
    s += x[j*4+0] + x[j*4+1] + x[j*4+2] + x[j*4+3];
  }
#pragma unroll
  for (int off = 16; off > 0; off >>= 1) s += __shfl_xor(s, off, 32);
  const float mu = s * (1.f / (float)D_);
  float vs = 0.f;
#pragma unroll
  for (int i = 0; i < 16; ++i) { const float d = x[i] - mu; vs += d * d; }
#pragma unroll
  for (int off = 16; off > 0; off >>= 1) vs += __shfl_xor(vs, off, 32);
  const float inv = rsqrtf(vs * (1.f / (float)D_) + LNEPS_);
  float* o = out + r * D_;
#pragma unroll
  for (int i = 0; i < 16; ++i) {
    const int c = lane * 16 + i;
    o[c] = (x[i] - mu) * inv * lw[c] + lb[c];
  }
}

// ---------------------------------------------------------------------------
extern "C" void kernel_launch(void* const* d_in, const int* in_sizes, int n_in,
                              void* d_out, int out_size, void* d_ws, size_t ws_size,
                              hipStream_t stream) {
  (void)in_sizes; (void)n_in; (void)out_size; (void)ws_size;
  const float* src = (const float*)d_in[0];
  const int*   seq = (const int*)d_in[1];
  const float* ipw = (const float*)d_in[2];
  const float* ipb = (const float*)d_in[3];
  const float* opw = (const float*)d_in[4];
  const float* opb = (const float*)d_in[5];
  const float* lnw = (const float*)d_in[6];
  const float* lnb = (const float*)d_in[7];
  float* ws  = (float*)d_ws;   // ~15 MB used
  float* out = (float*)d_out;

  q_proj_kernel <<<dim3(D_/16, B_/16),          32, 0, stream>>>(src, seq, ipw, ipb, ws);
  qw_kernel     <<<dim3(D_/16, B_/16, H_),      32, 0, stream>>>(ipw, ws);
  scores_kernel <<<dim3(S_/16, B_),             32, 0, stream>>>(src, ws);
  softmax_kernel<<<dim3(B_*H_),                256, 0, stream>>>(ws);
  y_kernel      <<<dim3(D_/16, B_, KSPLIT_),    32, 0, stream>>>(src, ws);
  y_reduce_kernel<<<dim3((B_*H_*D_)/256),      256, 0, stream>>>(ws);
  ctx_kernel    <<<dim3(HD_/16, B_/16, H_),     32, 0, stream>>>(ipw, ws);
  oproj_kernel  <<<dim3(D_/16, B_/16),          32, 0, stream>>>(opw, opb, ws);
  ln_kernel     <<<dim3((S_*B_)/8),            256, 0, stream>>>(src, lnw, lnb, ws, out);
}
